// MlmCls_45217415692301
// MI455X (gfx1250) — matrix operand order repaired and checked by reading, not compiled, not run on hardware
//
#include <hip/hip_runtime.h>
#include <hip/hip_bf16.h>

typedef __attribute__((ext_vector_type(16))) _Float16 v16h;
typedef __attribute__((ext_vector_type(8)))  _Float16 h8;
typedef __attribute__((ext_vector_type(4)))  _Float16 h4;
typedef __attribute__((ext_vector_type(8)))  float    v8f;
typedef __attribute__((ext_vector_type(4)))  float    f4;

#define B_SZ      32
#define S_SZ      128
#define M_TOK     (B_SZ * S_SZ)          // 4096 tokens
#define D_FEAT    768
#define NUM_TAG_  10000
#define NUM_VOC   21128
#define HIDDEN_   256
#define K_EMB     (2 * D_FEAT)           // 1536

#define KC   32                          // K chunk per WMMA step
#define LDH  (KC + 8)                    // LDS row stride in halfs (80B: keeps 16B alignment)

// ---- d_out layout (concatenated outputs, return order) ----
#define OUT_SIG  0
#define OUT_EMB  (B_SZ * NUM_TAG_)                 // 320000
#define OUT_LTAG (OUT_EMB + B_SZ * HIDDEN_)        // 328192
#define OUT_LMLM (OUT_LTAG + 1)                    // 328193

// ---- workspace layout (floats) ----
#define NT_MLM   ((NUM_VOC + 127) / 128)           // 166 vocab tiles
#define WS_PMAX  0
#define WS_PSUM  (WS_PMAX + NT_MLM * M_TOK)
#define WS_GOLD  (WS_PSUM + NT_MLM * M_TOK)
#define WS_EMB   (WS_GOLD + M_TOK)

// ---------- WMMA fragment loaders: two aligned 16B LDS loads each ----------
// A 16x32 f16: lane m=lane&15; halves 0..7 = K kb..kb+7, 8..15 = K kb+16..kb+23, kb=(lane>>4)*8
static __device__ inline v16h frag_a(const _Float16* As, int lane) {
  const int m  = lane & 15;
  const int kb = (lane >> 4) * 8;
  const _Float16* r = As + m * LDH + kb;          // byte offsets 0/16, rows 80B apart
  h8 lo = *(const h8*)(r);
  h8 hi = *(const h8*)(r + 16);
  return __builtin_shufflevector(lo, hi, 0,1,2,3,4,5,6,7,8,9,10,11,12,13,14,15);
}
// B 32x16 f16: lane col n=lane&15; halves 0..7 = K kb..kb+7, 8..15 = K kb+16..kb+23, kb=(lane>>4)*8
static __device__ inline v16h frag_b(const _Float16* Bs, int lane) {
  const int n  = lane & 15;
  const int kb = (lane >> 4) * 8;
  const _Float16* r = Bs + n * LDH + kb;          // byte offsets 0/16
  h8 lo = *(const h8*)(r);
  h8 hi = *(const h8*)(r + 16);
  return __builtin_shufflevector(lo, hi, 0,1,2,3,4,5,6,7,8,9,10,11,12,13,14,15);
}

static __device__ inline h4 cvt4(f4 v) {
  h4 h = { (_Float16)v.x, (_Float16)v.y, (_Float16)v.z, (_Float16)v.w };
  return h;
}

// =================== MLM head: [4096x768] @ [768x21128], double-buffered ===================
// grid = (166 n-tiles, 32 m-tiles), block = 256 (8 waves). Wave w owns M rows w*16..w*16+15,
// all 8 N-subtiles -> 8 f32 accumulators.
__launch_bounds__(256, 1)
__global__ void mlm_gemm(const float* __restrict__ title,   // [4096 x 768]
                         const float* __restrict__ Wm,      // [21128 x 768]
                         const float* __restrict__ bm,      // [21128]
                         const int*   __restrict__ lab,     // [4096]
                         float* __restrict__ pmax,          // [166][4096]
                         float* __restrict__ psum,          // [166][4096]
                         float* __restrict__ gold)          // [4096]
{
  __shared__ __align__(16) _Float16 As[2][128 * LDH];
  __shared__ __align__(16) _Float16 Bs[2][128 * LDH];
  const int tid  = threadIdx.x;
  const int lane = tid & 31;
  const int w    = tid >> 5;
  const int m0   = blockIdx.y * 128;
  const int n0   = blockIdx.x * 128;

  v8f acc[8];
#pragma unroll
  for (int s = 0; s < 8; ++s)
#pragma unroll
    for (int r = 0; r < 8; ++r) acc[s][r] = 0.f;

  // per-thread register staging: 128x32 tile = 1024 float4 / 256 threads = 4 each, per matrix
  f4 sa[4], sb[4];
  auto gload = [&](int k0) {
#pragma unroll
    for (int j = 0; j < 4; ++j) {
      int idx = tid + j * 256;                 // 0..1023
      int m = idx >> 3;                        // 0..127
      int k = (idx & 7) * 4;                   // 0..28
      sa[j] = *(const f4*)(title + (m0 + m) * D_FEAT + k0 + k);
      int v = n0 + m;                          // OOB rows: clamp (epilogue masks n>=NUM_VOC)
      v = (v < NUM_VOC) ? v : (NUM_VOC - 1);
      sb[j] = *(const f4*)(Wm + v * D_FEAT + k0 + k);
    }
  };
  auto sstore = [&](int buf) {
#pragma unroll
    for (int j = 0; j < 4; ++j) {
      int idx = tid + j * 256;
      int m = idx >> 3;
      int k = (idx & 7) * 4;
      *(h4*)(&As[buf][m * LDH + k]) = cvt4(sa[j]);
      *(h4*)(&Bs[buf][m * LDH + k]) = cvt4(sb[j]);
    }
  };

  gload(0);
  sstore(0);
  int cur = 0;
  const int NKC = D_FEAT / KC;                  // 24
  for (int kc = 0; kc < NKC; ++kc) {
    if (kc + 1 < NKC) gload((kc + 1) * KC);     // global loads in flight across barrier
    __syncthreads();                            // buf[cur] stores visible; buf[cur^1] reads done
    if (kc + 1 < NKC) sstore(cur ^ 1);          // fill next buffer while computing
    v16h a = frag_a(As[cur] + (w * 16) * LDH, lane);
    v16h bcur = frag_b(Bs[cur], lane);          // subtile 0
#pragma unroll
    for (int s = 0; s < 8; ++s) {               // B frags pipelined one subtile ahead
      v16h bnext = bcur;
      if (s < 7) bnext = frag_b(Bs[cur] + ((s + 1) * 16) * LDH, lane);
      acc[s] = __builtin_amdgcn_wmma_f32_16x16x32_f16(false, a, false, bcur,
                                                      (short)0, acc[s], false, false);
      bcur = bnext;
    }
    cur ^= 1;
  }

  // D layout: vgpr r, lanes0-15 -> M=r, lanes16-31 -> M=8+r; N = n0 + s*16 + (lane&15)
  const int half = lane >> 4;
  const int col  = lane & 15;

#pragma unroll
  for (int s = 0; s < 8; ++s) {                 // bias + OOB mask
    int n = n0 + s * 16 + col;
    bool ok = (n < NUM_VOC);
    float bias = ok ? bm[n] : 0.f;
#pragma unroll
    for (int r = 0; r < 8; ++r)
      acc[s][r] = ok ? (acc[s][r] + bias) : -__builtin_inff();
  }

  float rmax[8], rsum[8];
#pragma unroll
  for (int r = 0; r < 8; ++r) {                 // row max over this tile's 128 cols
    float t = acc[0][r];
#pragma unroll
    for (int s = 1; s < 8; ++s) t = fmaxf(t, acc[s][r]);
    t = fmaxf(t, __shfl_xor(t, 1));
    t = fmaxf(t, __shfl_xor(t, 2));
    t = fmaxf(t, __shfl_xor(t, 4));
    t = fmaxf(t, __shfl_xor(t, 8));
    rmax[r] = t;
  }
#pragma unroll
  for (int r = 0; r < 8; ++r) {                 // row sum of exp(x - rmax)
    float t = 0.f;
#pragma unroll
    for (int s = 0; s < 8; ++s) t += __expf(acc[s][r] - rmax[r]);
    t += __shfl_xor(t, 1);
    t += __shfl_xor(t, 2);
    t += __shfl_xor(t, 4);
    t += __shfl_xor(t, 8);
    rsum[r] = t;
  }

  if (col == 0) {
#pragma unroll
    for (int r = 0; r < 8; ++r) {
      int m = m0 + w * 16 + half * 8 + r;
      pmax[blockIdx.x * M_TOK + m] = rmax[r];
      psum[blockIdx.x * M_TOK + m] = rsum[r];
    }
  }

  int labv[8];                                   // gold-logit gather (statically unrolled)
#pragma unroll
  for (int r = 0; r < 8; ++r) labv[r] = lab[m0 + w * 16 + half * 8 + r];
#pragma unroll
  for (int s = 0; s < 8; ++s) {
    int n = n0 + s * 16 + col;
#pragma unroll
    for (int r = 0; r < 8; ++r)
      if (labv[r] == n) gold[m0 + w * 16 + half * 8 + r] = acc[s][r];
  }
}

// =================== token-wise merge of 166 partials -> masked CE sum ===================
__launch_bounds__(256, 1)
__global__ void mlm_reduce(const float* __restrict__ pmax,
                           const float* __restrict__ psum,
                           const float* __restrict__ gold,
                           const int*   __restrict__ lab,
                           float* __restrict__ loss_mlm)
{
  __shared__ float red[256];
  const int tok = blockIdx.x * 256 + threadIdx.x;
  float gmax = -__builtin_inff();
  for (int t = 0; t < NT_MLM; ++t) gmax = fmaxf(gmax, pmax[t * M_TOK + tok]);
  float s = 0.f;
  for (int t = 0; t < NT_MLM; ++t)
    s += psum[t * M_TOK + tok] * __expf(pmax[t * M_TOK + tok] - gmax);
  float ce = (gmax + __logf(s)) - gold[tok];
  red[threadIdx.x] = (lab[tok] != 0) ? ce : 0.f;
  __syncthreads();
  for (int off = 128; off > 0; off >>= 1) {
    if (threadIdx.x < off) red[threadIdx.x] += red[threadIdx.x + off];
    __syncthreads();
  }
  if (threadIdx.x == 0) atomicAdd(loss_mlm, red[0]);
}

// =================== Tag head: [32x768] @ [768x10000] + sigmoid + BCE sum ===================
// grid = 79 n-tiles, block = 256. Wave w: M-subtile = w>>2, two N-subtiles (w&3)*2+{0,1}.
__launch_bounds__(256, 1)
__global__ void tag_gemm(const float* __restrict__ video,      // [32 x 768]
                         const float* __restrict__ Wt,         // [10000 x 768]
                         const float* __restrict__ bt,
                         const float* __restrict__ label_tag,  // [32 x 10000]
                         float* __restrict__ out_sig,          // [32 x 10000]
                         float* __restrict__ loss_tag)
{
  __shared__ __align__(16) _Float16 As[32 * LDH];
  __shared__ __align__(16) _Float16 Bs[128 * LDH];
  __shared__ float red[256];
  const int tid = threadIdx.x, lane = tid & 31, w = tid >> 5;
  const int n0 = blockIdx.x * 128;
  const int msub = w >> 2, nbase = (w & 3) * 2;

  v8f acc[2];
#pragma unroll
  for (int j = 0; j < 2; ++j)
#pragma unroll
    for (int r = 0; r < 8; ++r) acc[j][r] = 0.f;

  for (int k0 = 0; k0 < D_FEAT; k0 += KC) {
    {                                            // A: 32x32 = 256 float4, 1 per thread
      int m = tid >> 3, k = (tid & 7) * 4;
      *(h4*)(&As[m * LDH + k]) = cvt4(*(const f4*)(video + m * D_FEAT + k0 + k));
    }
#pragma unroll
    for (int j = 0; j < 4; ++j) {                // B: 128x32 = 1024 float4, clamped rows
      int idx = tid + j * 256;
      int n = idx >> 3, k = (idx & 7) * 4;
      int v = n0 + n;
      v = (v < NUM_TAG_) ? v : (NUM_TAG_ - 1);   // epilogue skips n>=NUM_TAG_
      *(h4*)(&Bs[n * LDH + k]) = cvt4(*(const f4*)(Wt + v * D_FEAT + k0 + k));
    }
    __syncthreads();
    v16h a = frag_a(As + msub * 16 * LDH, lane);
    v16h b0 = frag_b(Bs + nbase * 16 * LDH, lane);
    v16h b1 = frag_b(Bs + (nbase + 1) * 16 * LDH, lane);
    acc[0] = __builtin_amdgcn_wmma_f32_16x16x32_f16(false, a, false, b0,
                                                    (short)0, acc[0], false, false);
    acc[1] = __builtin_amdgcn_wmma_f32_16x16x32_f16(false, a, false, b1,
                                                    (short)0, acc[1], false, false);
    __syncthreads();
  }

  const int half = lane >> 4, col = lane & 15;
  float lsum = 0.f;
#pragma unroll
  for (int j = 0; j < 2; ++j) {
    int n = n0 + (nbase + j) * 16 + col;
    if (n < NUM_TAG_) {
      float bias = bt[n];
#pragma unroll
      for (int r = 0; r < 8; ++r) {
        int m = msub * 16 + half * 8 + r;
        float p = acc[j][r] + bias;
        out_sig[m * NUM_TAG_ + n] = 1.f / (1.f + __expf(-p));
        float y = label_tag[m * NUM_TAG_ + n];
        lsum += fmaxf(p, 0.f) - p * y + log1pf(__expf(-fabsf(p)));
      }
    }
  }
  red[tid] = lsum;
  __syncthreads();
  for (int off = 128; off > 0; off >>= 1) {
    if (tid < off) red[tid] += red[tid + off];
    __syncthreads();
  }
  if (tid == 0) atomicAdd(loss_tag, red[0]);
}

// =================== Embedding head: [32x1536] @ [1536x256] (single workgroup) ===================
__launch_bounds__(256, 1)
__global__ void emb_gemm(const float* __restrict__ video,
                         const float* __restrict__ title,      // uses title[:,0,:]
                         const float* __restrict__ Wh,         // [256 x 1536]
                         const float* __restrict__ bh,
                         float* __restrict__ emb)              // [32 x 256] raw
{
  __shared__ __align__(16) _Float16 As[32 * LDH];
  __shared__ __align__(16) _Float16 Bs[256 * LDH];
  const int tid = threadIdx.x, lane = tid & 31, w = tid >> 5;
  const int msub = w >> 2, nbase = (w & 3) * 4;

  v8f acc[4];
#pragma unroll
  for (int j = 0; j < 4; ++j)
#pragma unroll
    for (int r = 0; r < 8; ++r) acc[j][r] = 0.f;

  for (int k0 = 0; k0 < K_EMB; k0 += KC) {
    {                                            // A = concat(video, title[:,0]), 1 float4/thread
      int m = tid >> 3, k = (tid & 7) * 4;
      int kk = k0 + k;                           // k0 multiple of 32 -> no f4 straddle at 768
      f4 av = (kk < D_FEAT) ? *(const f4*)(video + m * D_FEAT + kk)
                            : *(const f4*)(title + m * (S_SZ * D_FEAT) + (kk - D_FEAT));
      *(h4*)(&As[m * LDH + k]) = cvt4(av);
    }
#pragma unroll
    for (int j = 0; j < 8; ++j) {                // B: 256x32 = 2048 float4
      int idx = tid + j * 256;
      int n = idx >> 3, k = (idx & 7) * 4;
      *(h4*)(&Bs[n * LDH + k]) = cvt4(*(const f4*)(Wh + n * K_EMB + k0 + k));
    }
    __syncthreads();
    v16h a = frag_a(As + msub * 16 * LDH, lane);
    v16h bcur = frag_b(Bs + nbase * 16 * LDH, lane);
#pragma unroll
    for (int j = 0; j < 4; ++j) {
      v16h bnext = bcur;
      if (j < 3) bnext = frag_b(Bs + (nbase + j + 1) * 16 * LDH, lane);
      acc[j] = __builtin_amdgcn_wmma_f32_16x16x32_f16(false, a, false, bcur,
                                                      (short)0, acc[j], false, false);
      bcur = bnext;
    }
    __syncthreads();
  }

  const int half = lane >> 4, col = lane & 15;
#pragma unroll
  for (int j = 0; j < 4; ++j) {
    int n = (nbase + j) * 16 + col;
#pragma unroll
    for (int r = 0; r < 8; ++r) {
      int m = msub * 16 + half * 8 + r;
      emb[m * HIDDEN_ + n] = acc[j][r] + bh[n];
    }
  }
}

__global__ void emb_norm(const float* __restrict__ emb, float* __restrict__ out_emb) {
  __shared__ float red[256];
  const int row = blockIdx.x, tid = threadIdx.x;
  float v = emb[row * HIDDEN_ + tid];
  red[tid] = v * v;
  __syncthreads();
  for (int off = 128; off > 0; off >>= 1) {
    if (tid < off) red[tid] += red[tid + off];
    __syncthreads();
  }
  float nrm = sqrtf(red[0]);
  out_emb[row * HIDDEN_ + tid] = v / fmaxf(nrm, 1e-12f);
}

__global__ void zero_losses(float* out) {
  if (threadIdx.x == 0) { out[OUT_LTAG] = 0.f; out[OUT_LMLM] = 0.f; }
}
__global__ void finalize_tag(float* out) {
  if (threadIdx.x == 0) out[OUT_LTAG] *= (1.f / (float)(B_SZ * NUM_TAG_));
}

extern "C" void kernel_launch(void* const* d_in, const int* in_sizes, int n_in,
                              void* d_out, int out_size, void* d_ws, size_t ws_size,
                              hipStream_t stream) {
  (void)in_sizes; (void)n_in; (void)out_size; (void)ws_size;
  const float* video     = (const float*)d_in[0];
  const float* title     = (const float*)d_in[1];
  const float* label_tag = (const float*)d_in[2];
  const int*   label_mlm = (const int*)d_in[3];
  const float* Wt = (const float*)d_in[4];
  const float* bt = (const float*)d_in[5];
  const float* Wm = (const float*)d_in[6];
  const float* bm = (const float*)d_in[7];
  const float* Wh = (const float*)d_in[8];
  const float* bh = (const float*)d_in[9];
  float* out = (float*)d_out;
  float* ws  = (float*)d_ws;

  zero_losses<<<1, 32, 0, stream>>>(out);
  tag_gemm<<<dim3((NUM_TAG_ + 127) / 128), 256, 0, stream>>>(
      video, Wt, bt, label_tag, out + OUT_SIG, out + OUT_LTAG);
  finalize_tag<<<1, 32, 0, stream>>>(out);
  mlm_gemm<<<dim3(NT_MLM, M_TOK / 128), 256, 0, stream>>>(
      title, Wm, bm, label_mlm, ws + WS_PMAX, ws + WS_PSUM, ws + WS_GOLD);
  mlm_reduce<<<dim3(M_TOK / 256), 256, 0, stream>>>(
      ws + WS_PMAX, ws + WS_PSUM, ws + WS_GOLD, label_mlm, out + OUT_LMLM);
  emb_gemm<<<1, 256, 0, stream>>>(video, title, Wh, bh, ws + WS_EMB);
  emb_norm<<<B_SZ, HIDDEN_, 0, stream>>>(ws + WS_EMB, out + OUT_EMB);
}
